// LSTM_55937654063622
// MI455X (gfx1250) — matrix operand order repaired and checked by reading, not compiled, not run on hardware
//
#include <hip/hip_runtime.h>

#define B_  32
#define T_  2048
#define D_  512
#define H_  128
#define BT_ (B_*T_)

typedef __attribute__((ext_vector_type(16))) __bf16        v16bf;
typedef __attribute__((ext_vector_type(8)))  float         v8f;
typedef __attribute__((ext_vector_type(8)))  unsigned int  v8u;

union AView { v8u u; v16bf b; };

__device__ __forceinline__ unsigned short f2bf(float f) {
  unsigned u = __float_as_uint(f);
  unsigned r = u + 0x7FFFu + ((u >> 16) & 1u);
  return (unsigned short)(r >> 16);
}
__device__ __forceinline__ unsigned packbf(float lo, float hi) {
  return (unsigned)f2bf(lo) | ((unsigned)f2bf(hi) << 16);
}
__device__ __forceinline__ float sigm(float x) {
  return 1.0f / (1.0f + __expf(-x));
}

// ---------------------------------------------------------------------------
// Prep: swizzle W_cat / U_cat into per-lane WMMA B-operand layout (bf16 pairs),
// pack biases and LN params into contiguous [512] arrays.
// Wsw: [kt(16)][nt(32)][lane(32)][v(8)] uint ;  Usw: [nt*4+kk(128)][lane][v]
// ---------------------------------------------------------------------------
struct PrepPtrs {
  const float* W[4]; const float* U[4]; const float* b[4];
  const float* g[4]; const float* be[4];
};

__global__ __launch_bounds__(256) void prep_kernel(
    PrepPtrs p, unsigned* __restrict__ Wsw, unsigned* __restrict__ Usw,
    float* __restrict__ bcat, float* __restrict__ lng, float* __restrict__ lnb)
{
  int q = blockIdx.x * 256 + threadIdx.x;
  if (q < 131072) {                       // Wsw: 16*32*32*8
    int v = q & 7, lane = (q >> 3) & 31, nt = (q >> 8) & 31, kt = q >> 13;
    int hi = lane >> 4, lm = lane & 15;
    int n = nt * 16 + lm, gate = n >> 7, col = n & 127;
    int k = kt * 32 + ((v < 4) ? 2 * v : 16 + 2 * (v - 4)) + 8 * hi;
    const float* W = p.W[gate];
    Wsw[q] = packbf(W[k * 128 + col], W[(k + 1) * 128 + col]);
  } else if (q < 163840) {                // Usw: 128*32*8
    int q2 = q - 131072;
    int v = q2 & 7, lane = (q2 >> 3) & 31, tile = q2 >> 8;
    int hi = lane >> 4, lm = lane & 15;
    int kk = tile & 3, nt = tile >> 2;
    int k = kk * 32 + ((v < 4) ? 2 * v : 16 + 2 * (v - 4)) + 8 * hi;
    int n = nt * 16 + lm, gate = n >> 7, col = n & 127;
    const float* U = p.U[gate];
    Usw[q2] = packbf(U[k * 128 + col], U[(k + 1) * 128 + col]);
  } else if (q < 165376) {                // bcat / lng / lnb : 3*512
    int q3 = q - 163840;
    int which = q3 >> 9, n = q3 & 511;
    int gate = n >> 7, col = n & 127;
    if (which == 0)      bcat[n] = p.b[gate][col];
    else if (which == 1) lng[n]  = p.g[gate][col];
    else                 lnb[n]  = p.be[gate][col];
  }
}

// ---------------------------------------------------------------------------
// Phase 1: G[b*T+t][0..511] = x @ W_cat + b_cat   (bf16 WMMA, f32 accum)
// Each WG: 64 rows x full N=512. K loop: 16 chunks of 32. The next x chunk is
// staged across the barrier, and B tiles are software-pipelined 2 deep so
// WMMAs overlap the L2 loads instead of s_wait_loadcnt 0 per tile.
// ---------------------------------------------------------------------------
__global__ __launch_bounds__(256) void gemm_x_kernel(
    const float* __restrict__ x, const unsigned* __restrict__ Wsw,
    const float* __restrict__ bcat, float* __restrict__ G)
{
  __shared__ unsigned short a_sh[64][32];
  int tid = threadIdx.x, blk = blockIdx.x;
  int wv = tid >> 5, lane = tid & 31, hi = lane >> 4, lm = lane & 15;
  int mtile = wv >> 1, nhalf = wv & 1;

  v8f zero = {};
  v8f acc[16];
#pragma unroll
  for (int i = 0; i < 16; ++i) acc[i] = zero;

  int r = tid >> 2, cg = tid & 3;
  const float* xbase = x + ((size_t)blk * 64 + r) * 512 + cg * 8;
  float4 fa = *(const float4*)(xbase);
  float4 fb = *(const float4*)(xbase + 4);

  for (int kt = 0; kt < 16; ++kt) {
    {   // commit staged A chunk (bf16) to LDS
      unsigned short* d = &a_sh[r][cg * 8];
      d[0] = f2bf(fa.x); d[1] = f2bf(fa.y); d[2] = f2bf(fa.z); d[3] = f2bf(fa.w);
      d[4] = f2bf(fb.x); d[5] = f2bf(fb.y); d[6] = f2bf(fb.z); d[7] = f2bf(fb.w);
    }
    __syncthreads();
    float4 fa2 = fa, fb2 = fb;
    if (kt < 15) {      // prefetch next chunk; latency hidden by 16 WMMAs
      fa2 = *(const float4*)(xbase + (kt + 1) * 32);
      fb2 = *(const float4*)(xbase + (kt + 1) * 32 + 4);
    }
    AView a;
    int mrow = mtile * 16 + lm;
#pragma unroll
    for (int v = 0; v < 8; ++v) {
      int k0 = ((v < 4) ? 2 * v : 16 + 2 * (v - 4)) + 8 * hi;
      a.u[v] = *(const unsigned*)&a_sh[mrow][k0];
    }
    // B tiles for this kt: 16 consecutive 256-uint tiles starting at nhalf*16
    const unsigned* wbase = Wsw + ((size_t)(kt * 32 + nhalf * 16) * 32 + lane) * 8;
    v16bf bv0 = *(const v16bf*)(wbase);
    v16bf bv1 = *(const v16bf*)(wbase + 256);
#pragma unroll
    for (int ni = 0; ni < 16; ++ni) {
      v16bf bnext = bv1;
      if (ni < 14) bnext = *(const v16bf*)(wbase + (ni + 2) * 256);
      acc[ni] = __builtin_amdgcn_wmma_f32_16x16x32_bf16(
          false, a.b, false, bv0, (short)0, acc[ni], false, false);
      bv0 = bv1;
      bv1 = bnext;
    }
    __syncthreads();
    fa = fa2; fb = fb2;
  }
#pragma unroll
  for (int ni = 0; ni < 16; ++ni) {
    int nt = nhalf * 16 + ni;
#pragma unroll
    for (int i = 0; i < 8; ++i) {
      int m = mtile * 16 + i + hi * 8;
      int n = nt * 16 + lm;
      G[((size_t)blk * 64 + m) * 512 + n] = acc[ni][i] + bcat[n];
    }
  }
}

// ---------------------------------------------------------------------------
// Phase 2: sequential recurrence. 2 WGs x 512 threads (16 waves); WG wg owns
// batch rows [wg*16, wg*16+16). U_cat resident in LDS (swizzled B layout).
// Per step: WMMA (16x512 preact, G double-buffered in registers)
//           -> fused LN + sigmoid + zoneout update (one wave per batch row).
// LDS: u_lds 131072 | s_pre 32768 | h_f 8192 | c_f 8192 | h_bf 4096 | zm_f 8192
// ---------------------------------------------------------------------------
__global__ __launch_bounds__(512) void lstm_rec_kernel(
    const float* __restrict__ G, const unsigned* __restrict__ Usw,
    const float* __restrict__ lng_g, const float* __restrict__ lnb_g,
    const float* __restrict__ h0, const float* __restrict__ c0,
    const int* __restrict__ zm, float* __restrict__ out)
{
  extern __shared__ char smem[];
  unsigned*       u_lds = (unsigned*)(smem);
  float*          s_pre = (float*)(smem + 131072);
  float*          h_f   = (float*)(smem + 131072 + 32768);
  float*          c_f   = (float*)(smem + 131072 + 32768 + 8192);
  unsigned short* h_bf  = (unsigned short*)(smem + 131072 + 32768 + 16384);
  float*          zm_f  = (float*)(smem + 131072 + 32768 + 16384 + 4096);

  int tid = threadIdx.x;
  int wg = blockIdx.x;            // 0..1
  int b0 = wg * 16;
  int wv = tid >> 5, lane = tid & 31, hi = lane >> 4, lm = lane & 15;

  // ---- one-time init: U into LDS, mask into LDS, h0/c0 ----
  {
    const uint4* src = (const uint4*)Usw;
    uint4* dst = (uint4*)u_lds;
    for (int i = tid; i < 8192; i += 512) dst[i] = src[i];
  }
  for (int i = tid; i < 2048; i += 512) zm_f[i] = (float)zm[i];
  for (int i = tid; i < 2048; i += 512) {
    int rr = i >> 7, j = i & 127;
    float hv = h0[(b0 + rr) * 128 + j];
    float cv = c0[(b0 + rr) * 128 + j];
    h_f[i] = hv; c_f[i] = cv; h_bf[i] = f2bf(hv);
  }

  // ---- per-thread constants for fused LN/update stage ----
  // wave wv owns batch row `row`; lane owns columns j0..j0+3 of all 4 gates.
  int row = wv, j0 = lane * 4;
  float lg[16], lb[16];
#pragma unroll
  for (int g = 0; g < 4; ++g)
#pragma unroll
    for (int jj = 0; jj < 4; ++jj) {
      int n = g * 128 + j0 + jj;
      lg[g * 4 + jj] = lng_g[n];
      lb[g * 4 + jj] = lnb_g[n];
    }

  // ---- G double buffer: prologue load for t=0 ----
  float gcur[16], gnext[16];
#pragma unroll
  for (int ni = 0; ni < 2; ++ni)
#pragma unroll
    for (int i = 0; i < 8; ++i)
      gcur[ni * 8 + i] =
          G[((size_t)(b0 + i + hi * 8) * T_ + 0) * 512 + (ni * 16 + wv) * 16 + lm];

  __syncthreads();

  for (int t = 0; t < T_; ++t) {
    // ================= stage A: s_pre = G[:,t,:] + h @ U_cat =================
    AView av[4];
#pragma unroll
    for (int kk = 0; kk < 4; ++kk) {
#pragma unroll
      for (int v = 0; v < 8; ++v) {
        int k0 = ((v < 4) ? 2 * v : 16 + 2 * (v - 4)) + 8 * hi + kk * 32;
        av[kk].u[v] = *(const unsigned*)&h_bf[lm * 128 + k0];
      }
    }
    // issue next step's G loads early; consumed next iteration
    int tt = (t + 1 < T_) ? t + 1 : t;
#pragma unroll
    for (int ni = 0; ni < 2; ++ni)
#pragma unroll
      for (int i = 0; i < 8; ++i)
        gnext[ni * 8 + i] =
            G[((size_t)(b0 + i + hi * 8) * T_ + tt) * 512 + (ni * 16 + wv) * 16 + lm];

#pragma unroll
    for (int ni = 0; ni < 2; ++ni) {
      int nt = ni * 16 + wv;
      // clause all 4 B vectors from LDS before the serial WMMA chain
      v16bf bvs[4];
#pragma unroll
      for (int kk = 0; kk < 4; ++kk)
        bvs[kk] = *(const v16bf*)&u_lds[((nt * 4 + kk) * 32 + lane) * 8];
      v8f acc = {};
#pragma unroll
      for (int i = 0; i < 8; ++i) acc[i] = gcur[ni * 8 + i];
#pragma unroll
      for (int kk = 0; kk < 4; ++kk)
        acc = __builtin_amdgcn_wmma_f32_16x16x32_bf16(
            false, av[kk].b, false, bvs[kk], (short)0, acc, false, false);
#pragma unroll
      for (int i = 0; i < 8; ++i)
        s_pre[(i + hi * 8) * 512 + nt * 16 + lm] = acc[i];
    }
    __syncthreads();

    // ======= stage B: fused LayerNorm + sigmoid + zoneout update =======
    {
      float vals[16], sum[4], ss[4];
#pragma unroll
      for (int g = 0; g < 4; ++g) {
        float4 v4 = *(const float4*)&s_pre[row * 512 + g * 128 + j0];
        vals[g * 4 + 0] = v4.x; vals[g * 4 + 1] = v4.y;
        vals[g * 4 + 2] = v4.z; vals[g * 4 + 3] = v4.w;
        sum[g] = v4.x + v4.y + v4.z + v4.w;
        ss[g]  = v4.x * v4.x + v4.y * v4.y + v4.z * v4.z + v4.w * v4.w;
      }
#pragma unroll
      for (int m = 1; m < 32; m <<= 1) {
#pragma unroll
        for (int g = 0; g < 4; ++g) {
          sum[g] += __shfl_xor(sum[g], m, 32);
          ss[g]  += __shfl_xor(ss[g], m, 32);
        }
      }
      float act[16];
#pragma unroll
      for (int g = 0; g < 4; ++g) {
        float mean = sum[g] * (1.0f / 128.0f);
        float var  = ss[g] * (1.0f / 128.0f) - mean * mean;
        float rstd = rsqrtf(var + 1e-5f);
#pragma unroll
        for (int jj = 0; jj < 4; ++jj) {
          float a = (vals[g * 4 + jj] - mean) * rstd * lg[g * 4 + jj] + lb[g * 4 + jj];
          act[g * 4 + jj] = sigm(a);
        }
      }
      float mt = zm_f[t];
      float om = 1.0f - mt;
      int idx = row * 128 + j0;
      float4 c4 = *(const float4*)&c_f[idx];
      float4 h4 = *(const float4*)&h_f[idx];
      float cn[4], hn[4];
#pragma unroll
      for (int jj = 0; jj < 4; ++jj) {
        float co = (jj == 0) ? c4.x : (jj == 1) ? c4.y : (jj == 2) ? c4.z : c4.w;
        float ho = (jj == 0) ? h4.x : (jj == 1) ? h4.y : (jj == 2) ? h4.z : h4.w;
        float f  = act[0 * 4 + jj];
        float ig = act[1 * 4 + jj];
        float o  = act[2 * 4 + jj];
        float ch = act[3 * 4 + jj];
        cn[jj] = mt * co + om * (f * co + ig * ch);
        hn[jj] = mt * ho + om * (o * sigm(cn[jj]));
      }
      *(float4*)&c_f[idx] = make_float4(cn[0], cn[1], cn[2], cn[3]);
      *(float4*)&h_f[idx] = make_float4(hn[0], hn[1], hn[2], hn[3]);
      uint2 hb;
      hb.x = packbf(hn[0], hn[1]);
      hb.y = packbf(hn[2], hn[3]);
      *(uint2*)&h_bf[idx] = hb;
      *(float4*)&out[((size_t)(b0 + row) * T_ + t) * 128 + j0] =
          make_float4(hn[0], hn[1], hn[2], hn[3]);
    }
    // rotate G double buffer
#pragma unroll
    for (int i = 0; i < 16; ++i) gcur[i] = gnext[i];
    __syncthreads();
  }

  // final h_T appended after out_seq
  for (int i = tid; i < 2048; i += 512) {
    int rr = i >> 7, j = i & 127;
    out[(size_t)BT_ * 128 + (size_t)(b0 + rr) * 128 + j] = h_f[i];
  }
}

// ---------------------------------------------------------------------------
extern "C" void kernel_launch(void* const* d_in, const int* in_sizes, int n_in,
                              void* d_out, int out_size, void* d_ws, size_t ws_size,
                              hipStream_t stream)
{
  (void)in_sizes; (void)n_in; (void)out_size; (void)ws_size;

  const float* x  = (const float*)d_in[0];
  PrepPtrs p;
  for (int g = 0; g < 4; ++g) {
    p.W[g]  = (const float*)d_in[1 + g];
    p.U[g]  = (const float*)d_in[5 + g];
    p.b[g]  = (const float*)d_in[9 + g];
    p.g[g]  = (const float*)d_in[13 + 2 * g];
    p.be[g] = (const float*)d_in[14 + 2 * g];
  }
  const float* h0 = (const float*)d_in[21];
  const float* c0 = (const float*)d_in[22];
  const int*   zm = (const int*)d_in[23];
  float* out = (float*)d_out;

  char* ws = (char*)d_ws;
  float*    G    = (float*)ws;                                   // 134217728 B
  unsigned* Wsw  = (unsigned*)(ws + (size_t)134217728);          //    524288 B
  unsigned* Usw  = (unsigned*)(ws + (size_t)134217728 + 524288); //    131072 B
  float*    bcat = (float*)(ws + (size_t)134217728 + 524288 + 131072);
  float*    lng  = (float*)(ws + (size_t)134217728 + 524288 + 131072 + 2048);
  float*    lnb  = (float*)(ws + (size_t)134217728 + 524288 + 131072 + 4096);

  prep_kernel<<<646, 256, 0, stream>>>(p, Wsw, Usw, bcat, lng, lnb);
  gemm_x_kernel<<<BT_ / 64, 256, 0, stream>>>(x, Wsw, bcat, G);

  const size_t shmem = 188416;  // 184 KB dynamic LDS (within 320 KB/WGP)
  (void)hipFuncSetAttribute((const void*)lstm_rec_kernel,
                            hipFuncAttributeMaxDynamicSharedMemorySize, (int)shmem);
  lstm_rec_kernel<<<2, 512, shmem, stream>>>(G, Usw, lng, lnb, h0, c0, zm, out);
}
